// HGTModel_10660108829394
// MI455X (gfx1250) — matrix operand order repaired and checked by reading, not compiled, not run on hardware
//
#include <hip/hip_runtime.h>

// ---------------------------------------------------------------------------
// CDNA5 (gfx1250) HGT forward pass.
// Dense GEMMs: v_wmma_f32_16x16x32_bf16 (wave32, fp32 accumulate).
// KQV GEMM A-operand staging: global_load_async_to_lds_b128 + s_wait_asynccnt.
// Edge phase: VALU + f32 atomicAdd / order-preserving-u32 atomicMax.
// ---------------------------------------------------------------------------

typedef __attribute__((ext_vector_type(16))) __bf16   v16bf;
typedef __attribute__((ext_vector_type(8)))  float    v8f;
typedef __attribute__((ext_vector_type(8)))  unsigned u32x8;
typedef __attribute__((ext_vector_type(4)))  unsigned u32x4;

#define HID 128
#define OUTD 64

__device__ __forceinline__ float gelu_exact(float x) {
  return 0.5f * x * (1.0f + erff(x * 0.70710678118654752f));
}
// Monotone order-preserving float<->uint encoding: segment-max becomes a
// single u32 atomicMax (GLOBAL_ATOMIC_MAX_U32) instead of a CAS loop.
__device__ __forceinline__ unsigned enc_f32(float f) {
  unsigned u = __float_as_uint(f);
  return (u & 0x80000000u) ? ~u : (u | 0x80000000u);
}
__device__ __forceinline__ float dec_f32(unsigned e) {
  return __uint_as_float((e & 0x80000000u) ? (e & 0x7fffffffu) : ~e);
}

enum { EPI_RELU_BF16 = 0, EPI_KQV = 1, EPI_SKIP_BF16 = 2, EPI_F32 = 3 };

// ---------------------------------------------------------------------------
// Generic tiled GEMM:  O = epilogue(A[MxK] @ W[KxN] + bias)
// 256 threads = 8 waves; tile 128x64; K chunked by 32.
// Wave w owns rows [16w,16w+16) x 64 cols -> 4 wmma accumulators.
// LDS rows padded to 40 elems (80B = 20 banks: conflict-free, 16B aligned).
// ---------------------------------------------------------------------------
template <bool A_BF16, bool GELU_A, int EPI>
__global__ __launch_bounds__(256) void gemm_wmma_kernel(
    const void* __restrict__ A, const float* __restrict__ W,
    const float* __restrict__ bias,
    void* __restrict__ O0, void* __restrict__ O1, void* __restrict__ O2,
    const __bf16* __restrict__ skip_src, const float* __restrict__ skip_gate,
    int M, int N, int K, int fout, int l2fout) {
  constexpr int BM = 128, BN = 64, BK = 32, LDR = 40;  // 80B row stride
  __shared__ alignas(16) __bf16 As[BM][LDR];
  __shared__ alignas(16) __bf16 Bs[BN][LDR];

  const int tid  = threadIdx.x;
  const int lane = tid & 31, wave = tid >> 5;
  const int half = lane >> 4, l16 = lane & 15;
  const int m0 = blockIdx.x * BM;
  const int n0 = blockIdx.y * BN;

  v8f acc[4] = {};

  for (int k0 = 0; k0 < K; k0 += BK) {
    // ---- stage A tile ----
    if constexpr (A_BF16 && !GELU_A) {
      // Pure byte copy: CDNA5 async DMA, 16B chunks, bypasses VGPRs.
      // (Rows past M read adjacent workspace; they only feed C-rows that the
      //  bounds-guarded epilogue never stores.)
      const unsigned ldsA = (unsigned)(size_t)(&As[0][0]);
      unsigned long long gbase =
          (unsigned long long)(const char*)A + (unsigned)(m0 * K + k0) * 2u;
#pragma unroll
      for (int i = tid; i < (BM * BK) / 8; i += 256) {  // 512 16B chunks
        int r = i >> 2, ch = i & 3;
        unsigned loff = ldsA + (unsigned)(r * (LDR * 2) + ch * 16);
        unsigned long long ga =
            gbase + (unsigned long long)((unsigned)(r * K) * 2u + ch * 16u);
        asm volatile("global_load_async_to_lds_b128 %0, %1, off"
                     :: "v"(loff), "v"(ga) : "memory");
      }
    } else {
      if (k0 + BK < K)  // prefetch next K-chunk of this thread's A rows
        __builtin_prefetch((const float*)A +
                           (size_t)(m0 + (tid >> 1)) * K + k0 + BK +
                           ((tid & 1) << 4));
#pragma unroll
      for (int i = tid; i < BM * BK; i += 256) {
        int r = i >> 5, c = i & 31;
        int gr = m0 + r;
        float v = 0.f;
        if (gr < M) {
          if constexpr (A_BF16)
            v = (float)((const __bf16*)A)[(size_t)gr * K + (k0 + c)];
          else
            v = ((const float*)A)[(size_t)gr * K + (k0 + c)];
        }
        if constexpr (GELU_A) v = gelu_exact(v);
        As[r][c] = (__bf16)v;
      }
    }
    // ---- stage W tile transposed (f32 -> bf16): Bs[n][k] = W[k0+k][n0+n] ----
#pragma unroll
    for (int i = tid; i < BN * BK; i += 256) {
      int n = i >> 5, c = i & 31;
      Bs[n][c] = (__bf16)W[(size_t)(k0 + c) * N + (n0 + n)];
    }
    if constexpr (A_BF16 && !GELU_A)
      asm volatile("s_wait_asynccnt 0x0" ::: "memory");
    __syncthreads();

    // A fragment (16-bit A 16x32, contiguous K: lanes 0-15 K=0..15, 16-31 K=16..31),
    // loaded as 8 x b32 from LDS.
    v16bf afrag;
    {
      const unsigned* ar32 = (const unsigned*)&As[wave * 16 + l16][0];
      u32x8 aw;
#pragma unroll
      for (int v = 0; v < 8; ++v) {
        int kw = v + 8 * half;
        aw[v] = ar32[kw];
      }
      afrag = __builtin_bit_cast(v16bf, aw);
    }
#pragma unroll
    for (int t = 0; t < 4; ++t) {
      // B fragment (N across lanes; lanes 0-15 K=0..15, 16-31 K=16..31),
      // loaded as 2 x b128 from LDS.
      const u32x4* br = (const u32x4*)&Bs[t * 16 + l16][16 * half];
      u32x4 lo = br[0], hi = br[1];
      u32x8 bw;
#pragma unroll
      for (int v = 0; v < 4; ++v) { bw[v] = lo[v]; bw[4 + v] = hi[v]; }
      v16bf bfrag = __builtin_bit_cast(v16bf, bw);
      acc[t] = __builtin_amdgcn_wmma_f32_16x16x32_bf16(
          false, afrag, false, bfrag, (short)0, acc[t], false, false);
    }
    __syncthreads();
  }

  // ---- epilogue (all 32-bit addressing; block-uniform full-tile fast path) --
  float g = 0.f;
  if constexpr (EPI == EPI_SKIP_BF16) g = 1.f / (1.f + expf(-skip_gate[0]));
  const int row0 = m0 + wave * 16 + half * 8;

  auto epi_store = [&](int gr, int gc, float v) {
    if constexpr (EPI == EPI_RELU_BF16) {
      ((__bf16*)O0)[gr * N + gc] = (__bf16)fmaxf(v, 0.f);
    } else if constexpr (EPI == EPI_KQV) {
      int which = gc >> l2fout, cc = gc & (fout - 1);
      __bf16* dst = (which == 0) ? (__bf16*)O0
                                 : (which == 1) ? (__bf16*)O1 : (__bf16*)O2;
      dst[gr * fout + cc] = (__bf16)v;
    } else if constexpr (EPI == EPI_SKIP_BF16) {
      float hp = (float)skip_src[gr * N + gc];
      ((__bf16*)O0)[gr * N + gc] = (__bf16)(g * v + (1.f - g) * hp);
    } else {  // EPI_F32
      ((float*)O0)[gr * N + gc] = v;
    }
  };

  if (m0 + BM <= M) {  // uniform: no per-lane exec juggling
#pragma unroll
    for (int t = 0; t < 4; ++t) {
      const int gc = n0 + t * 16 + l16;
      const float bv = bias[gc];
#pragma unroll
      for (int r = 0; r < 8; ++r) epi_store(row0 + r, gc, acc[t][r] + bv);
    }
  } else {
#pragma unroll
    for (int t = 0; t < 4; ++t) {
      const int gc = n0 + t * 16 + l16;
      const float bv = bias[gc];
#pragma unroll
      for (int r = 0; r < 8; ++r)
        if (row0 + r < M) epi_store(row0 + r, gc, acc[t][r] + bv);
    }
  }
}

// ---------------------------------------------------------------------------
// Relation transform: dst[n, h*16+e] = sum_d src[n, h*16+d] * Wrel[h, d, e]
// One wave per (16-row block, head); single wmma, K zero-padded 16->32.
// ---------------------------------------------------------------------------
__global__ __launch_bounds__(128) void rel_gemm_kernel(
    const __bf16* __restrict__ src, const float* __restrict__ Wrel,
    float* __restrict__ dst, int Ns, int H, int fout) {
  const int lane = threadIdx.x & 31, wave = threadIdx.x >> 5;
  const int h = blockIdx.y * 4 + wave;
  if (h >= H) return;  // wave-uniform: EXEC stays all-ones for wmma
  const int half = lane >> 4, l16 = lane & 15;
  const int m0 = blockIdx.x * 16;
  const int gr = m0 + l16;

  v16bf a, b;
#pragma unroll
  for (int v = 0; v < 8; ++v) {
    int kk = (v < 4 ? 2 * v : 16 + 2 * (v - 4)) + 8 * half;  // A K-pattern
#pragma unroll
    for (int p = 0; p < 2; ++p) {
      int ka = kk + p;
      int kb = kk + p;  // B K-pattern
      __bf16 av = (__bf16)0.f, bv = (__bf16)0.f;
      if (ka < 16 && gr < Ns) av = src[gr * fout + h * 16 + ka];
      if (kb < 16) bv = (__bf16)Wrel[h * 256 + kb * 16 + l16];
      a[2 * v + p] = av;
      b[2 * v + p] = bv;
    }
  }
  v8f acc = {};
  acc = __builtin_amdgcn_wmma_f32_16x16x32_bf16(false, a, false, b, (short)0,
                                                acc, false, false);
  if (m0 + 16 <= Ns) {
#pragma unroll
    for (int r = 0; r < 8; ++r)
      dst[(m0 + half * 8 + r) * fout + h * 16 + l16] = acc[r];
  } else {
#pragma unroll
    for (int r = 0; r < 8; ++r) {
      int row = m0 + half * 8 + r;
      if (row < Ns) dst[row * fout + h * 16 + l16] = acc[r];
    }
  }
}

// ---------------------------------------------------------------------------
// Edge kernels (segment softmax over destination nodes); 32-bit indexing,
// H and fout are powers of two -> shifts instead of divides.
// ---------------------------------------------------------------------------
__global__ void score_kernel(const int* __restrict__ ei, int E, int EH,
                             const __bf16* __restrict__ q,
                             const float* __restrict__ krel,
                             const float* __restrict__ prel,
                             float* __restrict__ sc,
                             unsigned* __restrict__ menc, int dstBase, int l2H,
                             int fout) {
  int i = blockIdx.x * 256 + threadIdx.x;
  if (i >= EH) return;
  int e = i >> l2H, h = i & ((1 << l2H) - 1);
  int s = ei[e], d = ei[E + e];
  const __bf16* qr = q + d * fout + h * 16;
  const float* kr = krel + s * fout + h * 16;
  float acc = 0.f;
#pragma unroll
  for (int j = 0; j < 16; ++j) acc += (float)qr[j] * kr[j];
  float scv = acc * prel[h] * 0.25f;  // 1/sqrt(dh), dh = 16 in both layers
  sc[i] = scv;
  atomicMax(&menc[((dstBase + d) << l2H) + h], enc_f32(scv));
}

__global__ void expden_kernel(const int* __restrict__ ei, int E, int EH,
                              float* __restrict__ sc,
                              const unsigned* __restrict__ menc,
                              float* __restrict__ den, int dstBase, int l2H) {
  int i = blockIdx.x * 256 + threadIdx.x;
  if (i >= EH) return;
  int e = i >> l2H, h = i & ((1 << l2H) - 1);
  int d = ei[E + e];
  float m = dec_f32(menc[((dstBase + d) << l2H) + h]);
  float ex = expf(sc[i] - m);
  sc[i] = ex;  // overwrite score with exp
  atomicAdd(&den[((dstBase + d) << l2H) + h], ex);
}

__global__ void agg_kernel(const int* __restrict__ ei, int E, int EF,
                           const float* __restrict__ sc,
                           const float* __restrict__ den,
                           const float* __restrict__ vrel,
                           float* __restrict__ agg, int dstBase, int l2H,
                           int l2fout) {
  int i = blockIdx.x * 256 + threadIdx.x;
  if (i >= EF) return;
  int e = i >> l2fout, c = i & ((1 << l2fout) - 1);
  int h = c >> 4;  // dh = 16
  int s = ei[e], d = ei[E + e];
  float alpha = sc[(e << l2H) + h] /
                fmaxf(den[((dstBase + d) << l2H) + h], 1e-16f);
  atomicAdd(&agg[((dstBase + d) << l2fout) + c],
            vrel[(s << l2fout) + c] * alpha);
}

// ---------------------------------------------------------------------------
// Host orchestration
// ---------------------------------------------------------------------------
extern "C" void kernel_launch(void* const* d_in, const int* in_sizes, int n_in,
                              void* d_out, int out_size, void* d_ws,
                              size_t ws_size, hipStream_t stream) {
  (void)in_sizes; (void)n_in; (void)out_size; (void)ws_size;

  const float* x[3] = {(const float*)d_in[0], (const float*)d_in[1],
                       (const float*)d_in[2]};
  const int* ei[5] = {(const int*)d_in[3], (const int*)d_in[4],
                      (const int*)d_in[5], (const int*)d_in[6],
                      (const int*)d_in[7]};
  const float* lin_w[3] = {(const float*)d_in[8], (const float*)d_in[10],
                           (const float*)d_in[12]};
  const float* lin_b[3] = {(const float*)d_in[9], (const float*)d_in[11],
                           (const float*)d_in[13]};
  const float* kqv_w[2] = {(const float*)d_in[14], (const float*)d_in[22]};
  const float* kqv_b[2] = {(const float*)d_in[15], (const float*)d_in[23]};
  const float* krelW[2] = {(const float*)d_in[16], (const float*)d_in[24]};
  const float* vrelW[2] = {(const float*)d_in[17], (const float*)d_in[25]};
  const float* prelW[2] = {(const float*)d_in[18], (const float*)d_in[26]};
  const float* out_w[2] = {(const float*)d_in[19], (const float*)d_in[27]};
  const float* out_b[2] = {(const float*)d_in[20], (const float*)d_in[28]};
  const float* skipp[2] = {(const float*)d_in[21], (const float*)d_in[29]};

  const int Nn[3]   = {100000, 50000, 5000};
  const int base[3] = {0, 100000, 150000};
  const int NTOT    = 155000;
  const int IN_D[3] = {256, 128, 64};
  const int ES[5] = {1, 0, 0, 1, 2};
  const int ED[5] = {0, 0, 1, 2, 1};
  const int EE[5] = {250000, 250000, 200000, 100000, 100000};

  // ---- carve workspace (~330 MB) ----
  size_t off = 0;
  auto carve = [&](size_t bytes) -> void* {
    void* p = (char*)d_ws + off;
    off += (bytes + 255) & ~(size_t)255;
    return p;
  };
  __bf16* qb    = (__bf16*)carve((size_t)NTOT * HID * 2);
  __bf16* kb    = (__bf16*)carve((size_t)NTOT * HID * 2);
  __bf16* vb    = (__bf16*)carve((size_t)NTOT * HID * 2);
  __bf16* actB  = (__bf16*)carve((size_t)NTOT * HID * 2);   // h, then x1
  float*  relb  = (float*)carve((size_t)100000 * HID * 4);  // per-edge k/v_rel
  float*  scb   = (float*)carve((size_t)900000 * 8 * 4);
  unsigned* menc = (unsigned*)carve((size_t)NTOT * 8 * 4);
  float*  den   = (float*)carve((size_t)NTOT * 8 * 4);
  float*  agg   = (float*)carve((size_t)NTOT * HID * 4);

  // ---- input projections: h = relu(x @ lin_w + b), bf16 out ----
  for (int t = 0; t < 3; ++t) {
    dim3 grid((Nn[t] + 127) / 128, HID / 64);
    gemm_wmma_kernel<false, false, EPI_RELU_BF16><<<grid, 256, 0, stream>>>(
        x[t], lin_w[t], lin_b[t], actB + (size_t)base[t] * HID, nullptr,
        nullptr, nullptr, nullptr, Nn[t], HID, IN_D[t], 0, 0);
  }

  for (int L = 0; L < 2; ++L) {
    const int H = (L == 0) ? 8 : 4;
    const int l2H = (L == 0) ? 3 : 2;
    const int fout = (L == 0) ? HID : OUTD;
    const int l2fout = (L == 0) ? 7 : 6;
    const int kqvN = 3 * fout;

    // ---- KQV projections (async-LDS A staging), split into k/q/v bf16 ----
    for (int t = 0; t < 3; ++t) {
      dim3 grid((Nn[t] + 127) / 128, kqvN / 64);
      gemm_wmma_kernel<true, false, EPI_KQV><<<grid, 256, 0, stream>>>(
          actB + (size_t)base[t] * HID, kqv_w[L] + (size_t)t * HID * kqvN,
          kqv_b[L] + (size_t)t * kqvN, kb + (size_t)base[t] * fout,
          qb + (size_t)base[t] * fout, vb + (size_t)base[t] * fout, nullptr,
          nullptr, Nn[t], kqvN, HID, fout, l2fout);
    }

    (void)hipMemsetAsync(menc, 0, (size_t)NTOT * H * 4, stream);
    (void)hipMemsetAsync(den, 0, (size_t)NTOT * H * 4, stream);
    (void)hipMemsetAsync(agg, 0, (size_t)NTOT * fout * 4, stream);

    size_t scOff[5], sacc = 0;
    for (int e = 0; e < 5; ++e) { scOff[e] = sacc; sacc += (size_t)EE[e] * H; }

    // ---- pass 1: k_rel per edge type, scores + segment max ----
    for (int e = 0; e < 5; ++e) {
      int s = ES[e], d = ED[e];
      dim3 gr((Nn[s] + 15) / 16, (H + 3) / 4);
      rel_gemm_kernel<<<gr, 128, 0, stream>>>(
          kb + (size_t)base[s] * fout, krelW[L] + (size_t)e * H * 256, relb,
          Nn[s], H, fout);
      int EH = EE[e] * H;
      score_kernel<<<dim3((EH + 255) / 256), 256, 0, stream>>>(
          ei[e], EE[e], EH, qb + (size_t)base[d] * fout, relb,
          prelW[L] + e * H, scb + scOff[e], menc, base[d], l2H, fout);
    }
    // ---- pass 2: exp(sc - max), denominators ----
    for (int e = 0; e < 5; ++e) {
      int EH = EE[e] * H;
      expden_kernel<<<dim3((EH + 255) / 256), 256, 0, stream>>>(
          ei[e], EE[e], EH, scb + scOff[e], menc, den, base[ED[e]], l2H);
    }
    // ---- pass 3: v_rel per edge type, weighted aggregation ----
    for (int e = 0; e < 5; ++e) {
      int s = ES[e], d = ED[e];
      dim3 gr((Nn[s] + 15) / 16, (H + 3) / 4);
      rel_gemm_kernel<<<gr, 128, 0, stream>>>(
          vb + (size_t)base[s] * fout, vrelW[L] + (size_t)e * H * 256, relb,
          Nn[s], H, fout);
      int EF = EE[e] * fout;
      agg_kernel<<<dim3((EF + 255) / 256), 256, 0, stream>>>(
          ei[e], EE[e], EF, scb + scOff[e], den, relb, agg, base[d], l2H,
          l2fout);
    }
    // ---- output projection: gelu(agg) @ out_w + b (+ gated skip, L0) ----
    for (int t = 0; t < 3; ++t) {
      dim3 grid((Nn[t] + 127) / 128, fout / 64);
      if (L == 0) {
        gemm_wmma_kernel<false, true, EPI_SKIP_BF16><<<grid, 256, 0, stream>>>(
            agg + (size_t)base[t] * fout, out_w[L] + (size_t)t * fout * fout,
            out_b[L] + (size_t)t * fout, actB + (size_t)base[t] * HID, nullptr,
            nullptr, actB + (size_t)base[t] * HID, skipp[L] + t, Nn[t], fout,
            fout, 0, 0);
      } else {
        gemm_wmma_kernel<false, true, EPI_F32><<<grid, 256, 0, stream>>>(
            agg + (size_t)base[t] * fout, out_w[L] + (size_t)t * fout * fout,
            out_b[L] + (size_t)t * fout, (float*)d_out + (size_t)base[t] * OUTD,
            nullptr, nullptr, nullptr, nullptr, Nn[t], fout, fout, 0, 0);
      }
    }
  }
}
